// MambaLay_10075993276869
// MI455X (gfx1250) — matrix operand order repaired and checked by reading, not compiled, not run on hardware
//
#include <hip/hip_runtime.h>
#include <math.h>

typedef __attribute__((ext_vector_type(16))) _Float16 v16h;
typedef __attribute__((ext_vector_type(8)))  _Float16 v8h;
typedef __attribute__((ext_vector_type(2)))  _Float16 v2h;
typedef __attribute__((ext_vector_type(8)))  float    v8f;

#define B_    2
#define L_    2048
#define D_    256
#define DIN_  512
#define DTR_  128
#define NS_   32
#define NROWS (B_*L_)      /* 4096 */
#define CHUNKS 64
#define CLEN   32          /* CHUNKS*CLEN == L_ */
#define LPAD  40           /* f16 row stride in LDS: 80B, 16B-aligned, bank-spread */

__device__ __forceinline__ float sigmoidf_(float x){ return 1.f/(1.f+__expf(-x)); }
__device__ __forceinline__ float softplusf_(float x){ return (x>20.f)?x:log1pf(__expf(x)); }
__device__ __forceinline__ float geluf_(float x){ return 0.5f*x*(1.f+erff(x*0.70710678118654752f)); }

/* ---------------- LayerNorm over D=256, one row per block (256 thr) -------- */
__global__ void ln_kernel(const float* __restrict__ x, const float* __restrict__ g,
                          const float* __restrict__ be, float* __restrict__ h) {
  int row = blockIdx.x;
  int t   = threadIdx.x;
  float v = x[(size_t)row*D_ + t];
  float s = v, s2 = v*v;
  #pragma unroll
  for (int o = 16; o > 0; o >>= 1) { s += __shfl_down(s, o, 32); s2 += __shfl_down(s2, o, 32); }
  __shared__ float rs[8], rq[8];
  if ((t & 31) == 0) { rs[t>>5] = s; rq[t>>5] = s2; }
  __syncthreads();
  float tot = 0.f, tot2 = 0.f;
  #pragma unroll
  for (int i = 0; i < 8; i++) { tot += rs[i]; tot2 += rq[i]; }
  float mu  = tot  * (1.f/D_);
  float var = tot2 * (1.f/D_) - mu*mu;
  float inv = rsqrtf(var + 1e-5f);
  h[(size_t)row*D_ + t] = (v - mu)*inv*g[t] + be[t];
}

/* ---------------- LDS-staged, double-buffered, pipelined WMMA GEMM ----------
   Block (256 thr, 8 waves) computes 128 x (16*TN) tile; compile-time dims.
   Pipeline per K-step: cvt+ds_store slab (regs->LDS buf), barrier, issue next
   slab's global loads, compute TN x v_wmma from LDS while loads are in flight.
   MODE 0: plain  1: softplus(acc+bias)  2: acc+bias+resid  3: gelu(acc+bias) */
template<int MODE, int TN, int N, int K, int LDA, int LDB, int LDC>
__global__ void gemm16_wmma(const float* __restrict__ A, const float* __restrict__ Bw,
                            float* __restrict__ Dst,
                            const float* __restrict__ bias, const float* __restrict__ resid)
{
  constexpr int NB = 16*TN;
  constexpr int stripsN = N / NB;
  constexpr int NK = K / 32;
  __shared__ _Float16 Alds[2][128*LPAD];
  __shared__ _Float16 Blds[2][NB*LPAD];

  int tid   = threadIdx.x;
  int mblk  = blockIdx.x / stripsN;
  int nstrip= blockIdx.x - mblk*stripsN;
  int mBase = mblk*128;
  int n0    = nstrip*NB;

  int lane = tid & 31;
  int wv   = tid >> 5;
  int hl   = lane >> 4, r = lane & 15;

  v8f acc[TN];
  #pragma unroll
  for (int t = 0; t < TN; t++) acc[t] = (v8f){0.f,0.f,0.f,0.f,0.f,0.f,0.f,0.f};

  float2 aR[8];
  float2 bR[TN];

  auto gload = [&](int k0) {
    #pragma unroll
    for (int it = 0; it < 8; ++it) {                 /* A slab 128x32, b64 coalesced */
      int flat = it*256 + tid;
      int row = flat >> 4, kp = flat & 15;
      aR[it] = *(const float2*)(A + (size_t)(mBase + row)*LDA + k0 + 2*kp);
    }
    #pragma unroll
    for (int it = 0; it < TN; ++it) {                /* B slab 32xNB, coalesced over col */
      int flat = it*256 + tid;
      int col = flat % NB, kp = flat / NB;
      bR[it].x = Bw[(size_t)(k0 + 2*kp  )*LDB + n0 + col];
      bR[it].y = Bw[(size_t)(k0 + 2*kp+1)*LDB + n0 + col];
    }
  };
  auto lstore = [&](int buf) {
    #pragma unroll
    for (int it = 0; it < 8; ++it) {
      int flat = it*256 + tid;
      int row = flat >> 4, kp = flat & 15;
      v2h p; p[0] = (_Float16)aR[it].x; p[1] = (_Float16)aR[it].y;
      *(v2h*)(&Alds[buf][row*LPAD + 2*kp]) = p;
    }
    #pragma unroll
    for (int it = 0; it < TN; ++it) {
      int flat = it*256 + tid;
      int col = flat % NB, kp = flat / NB;
      v2h p; p[0] = (_Float16)bR[it].x; p[1] = (_Float16)bR[it].y;
      *(v2h*)(&Blds[buf][col*LPAD + 2*kp]) = p;
    }
  };

  gload(0);
  for (int ks = 0; ks < NK; ++ks) {
    int buf = ks & 1;
    lstore(buf);
    __syncthreads();
    if (ks + 1 < NK) gload((ks+1)*32);   /* in flight during WMMAs below */

    /* A fragment: row = wv*16 + r; runs at k = hl*8 and k = 16 + hl*8 */
    const _Float16* ab = &Alds[buf][(wv*16 + r)*LPAD];
    v8h alo = *(const v8h*)(ab + hl*8);
    v8h ahi = *(const v8h*)(ab + 16 + hl*8);
    v16h af = __builtin_shufflevector(alo, ahi, 0,1,2,3,4,5,6,7,8,9,10,11,12,13,14,15);

    v16h bf[TN];
    #pragma unroll
    for (int t = 0; t < TN; t++) {       /* all B fragments before the WMMA chain */
      const _Float16* bb = &Blds[buf][(t*16 + r)*LPAD];
      v8h blo = *(const v8h*)(bb + hl*8);
      v8h bhi = *(const v8h*)(bb + 16 + hl*8);
      bf[t] = __builtin_shufflevector(blo, bhi, 0,1,2,3,4,5,6,7,8,9,10,11,12,13,14,15);
    }
    #pragma unroll
    for (int t = 0; t < TN; t++)
      acc[t] = __builtin_amdgcn_wmma_f32_16x16x32_f16(false, af, false, bf[t],
                                                      (short)0, acc[t], false, false);
  }

  #pragma unroll
  for (int t = 0; t < TN; t++) {
    int col = n0 + t*16 + r;
    #pragma unroll
    for (int v = 0; v < 8; v++) {
      int row = mBase + wv*16 + v + hl*8;   /* D layout: M = vgpr + 8*half */
      float val = acc[t][v];
      if      (MODE == 1) val = softplusf_(val + bias[col]);
      else if (MODE == 2) val = val + bias[col] + resid[(size_t)row*LDC + col];
      else if (MODE == 3) val = geluf_(val + bias[col]);
      Dst[(size_t)row*LDC + col] = val;
    }
  }
}

/* ---------------- causal depthwise conv (K=4) + SiLU ----------------------- */
__global__ void conv_silu_kernel(const float* __restrict__ xr, const float* __restrict__ cw,
                                 const float* __restrict__ cb, float* __restrict__ xi) {
  int idx = blockIdx.x*blockDim.x + threadIdx.x;       /* B*L*DIN */
  int c  = idx & (DIN_-1);
  int bl = idx >> 9;
  int l  = bl & (L_-1);
  int b  = bl >> 11;
  float acc = cb[c];
  #pragma unroll
  for (int i = 0; i < 4; i++) {
    int ls = l - 3 + i;
    if (ls >= 0) acc += xr[((size_t)(b*L_ + ls))*1024 + c] * cw[i*DIN_ + c];
  }
  xi[idx] = acc * sigmoidf_(acc);
}

/* ---------------- chunked selective scan, phase A: per-chunk states -------- */
__global__ void scan_phaseA(const float* __restrict__ delta, const float* __restrict__ xi,
                            const float* __restrict__ xdbl,  const float* __restrict__ A_log,
                            float* __restrict__ S, float* __restrict__ P) {
  int d  = (blockIdx.x & 1)*256 + threadIdx.x;         /* channel 0..511 */
  int bc = blockIdx.x >> 1;                            /* b*CHUNKS + c   */
  int c  = bc & (CHUNKS-1);
  int b  = bc >> 6;
  float Ar[NS_], s[NS_], sum[NS_];
  #pragma unroll
  for (int n = 0; n < NS_; n++) { Ar[n] = -__expf(A_log[d*NS_ + n]); s[n] = 0.f; sum[n] = 0.f; }
  int l0 = c*CLEN;
  for (int j = 0; j < CLEN; ++j) {
    size_t rbl = (size_t)(b*L_ + l0 + j);
    float dt = delta[rbl*DIN_ + d];
    float u  = xi[rbl*DIN_ + d];
    const float* Brow = xdbl + rbl*192 + DTR_;         /* wave-uniform -> scalar loads */
    float du = dt*u;
    #pragma unroll
    for (int n = 0; n < NS_; n++) {
      float dA = dt*Ar[n];
      sum[n] += dA;
      s[n] = __expf(dA)*s[n] + du*Brow[n];
    }
  }
  size_t o = (((size_t)bc)*DIN_ + d)*NS_;
  #pragma unroll
  for (int n = 0; n < NS_; n++) { S[o+n] = s[n]; P[o+n] = __expf(sum[n]); }
}

/* ---------------- phase B: sequential stitch across chunks ----------------- */
__global__ void scan_phaseB(const float* __restrict__ S, const float* __restrict__ P,
                            float* __restrict__ Ti) {
  int idx = blockIdx.x*blockDim.x + threadIdx.x;       /* B*DIN*NS = 32768 */
  int n = idx & (NS_-1);
  int d = (idx >> 5) & (DIN_-1);
  int b = idx >> 14;
  float T = 0.f;
  for (int c = 0; c < CHUNKS; c++) {
    size_t o = (((size_t)(b*CHUNKS + c))*DIN_ + d)*NS_ + n;
    float sn = S[o] + P[o]*T;
    Ti[o] = T;                                         /* init state for chunk c */
    T = sn;
  }
}

/* ---------------- phase C: replay with true init, emit y*silu(res) --------- */
__global__ void scan_phaseC(const float* __restrict__ delta, const float* __restrict__ xi,
                            const float* __restrict__ xdbl,  const float* __restrict__ A_log,
                            const float* __restrict__ Dp,    const float* __restrict__ Ti,
                            const float* __restrict__ xr,    float* __restrict__ y) {
  int d  = (blockIdx.x & 1)*256 + threadIdx.x;
  int bc = blockIdx.x >> 1;
  int c  = bc & (CHUNKS-1);
  int b  = bc >> 6;
  float Ar[NS_], s[NS_];
  size_t o = (((size_t)bc)*DIN_ + d)*NS_;
  #pragma unroll
  for (int n = 0; n < NS_; n++) { Ar[n] = -__expf(A_log[d*NS_ + n]); s[n] = Ti[o+n]; }
  float dpar = Dp[d];
  int l0 = c*CLEN;
  for (int j = 0; j < CLEN; ++j) {
    size_t rbl = (size_t)(b*L_ + l0 + j);
    float dt = delta[rbl*DIN_ + d];
    float u  = xi[rbl*DIN_ + d];
    const float* Brow = xdbl + rbl*192 + DTR_;
    const float* Crow = xdbl + rbl*192 + DTR_ + NS_;
    float du = dt*u;
    float accy = 0.f;
    #pragma unroll
    for (int n = 0; n < NS_; n++) {
      s[n] = __expf(dt*Ar[n])*s[n] + du*Brow[n];
      accy += s[n]*Crow[n];
    }
    float yv = accy + u*dpar;
    float r  = xr[rbl*1024 + DIN_ + d];                /* res */
    yv *= r * sigmoidf_(r);
    y[rbl*DIN_ + d] = yv;
  }
}

/* ============================ host side ==================================== */
extern "C" void kernel_launch(void* const* d_in, const int* in_sizes, int n_in,
                              void* d_out, int out_size, void* d_ws, size_t ws_size,
                              hipStream_t stream) {
  (void)in_sizes; (void)n_in; (void)out_size; (void)ws_size;
  const float* x      = (const float*)d_in[0];
  const float* g      = (const float*)d_in[1];
  const float* be     = (const float*)d_in[2];
  const float* W_in   = (const float*)d_in[3];
  const float* conv_w = (const float*)d_in[4];
  const float* conv_b = (const float*)d_in[5];
  const float* W_x    = (const float*)d_in[6];
  const float* W_dt   = (const float*)d_in[7];
  const float* b_dt   = (const float*)d_in[8];
  const float* A_log  = (const float*)d_in[9];
  const float* D_par  = (const float*)d_in[10];
  const float* W_out  = (const float*)d_in[11];
  const float* b_out  = (const float*)d_in[12];
  const float* W_dns  = (const float*)d_in[13];
  const float* b_dns  = (const float*)d_in[14];
  float* out = (float*)d_out;
  float* ws  = (float*)d_ws;

  /* workspace layout (floats) */
  float* h    = ws;                 /* 4096*256   */
  float* xr   = ws +  1048576;      /* 4096*1024  */
  float* xi   = ws +  5242880;      /* 4096*512   */
  float* xdbl = ws +  7340032;      /* 4096*192   */
  float* dlt  = ws +  8126464;      /* 4096*512   */
  float* yv   = ws + 10223616;      /* 4096*512   */
  float* h2   = ws + 12320768;      /* 4096*256   */
  float* S    = ws + 13369344;      /* 2*64*512*32 */
  float* P    = ws + 15466496;      /* same       */
  float* Ti   = ws + 17563648;      /* same       */

  ln_kernel<<<NROWS, 256, 0, stream>>>(x, g, be, h);

  /* x_and_res = h @ W_in : 128x64 block tiles */
  gemm16_wmma<0,4, 1024,256, 256,1024,1024>
      <<<(NROWS/128)*(1024/64), 256, 0, stream>>>(h, W_in, xr, nullptr, nullptr);

  conv_silu_kernel<<<(NROWS*DIN_)/256, 256, 0, stream>>>(xr, conv_w, conv_b, xi);

  /* x_dbl = xi @ W_x */
  gemm16_wmma<0,4, 192,512, 512,192,192>
      <<<(NROWS/128)*(192/64), 256, 0, stream>>>(xi, W_x, xdbl, nullptr, nullptr);

  /* delta = softplus(x_dbl[:, :128] @ W_dt + b_dt) */
  gemm16_wmma<1,4, 512,128, 192,512,512>
      <<<(NROWS/128)*(512/64), 256, 0, stream>>>(xdbl, W_dt, dlt, b_dt, nullptr);

  scan_phaseA<<<B_*CHUNKS*2, 256, 0, stream>>>(dlt, xi, xdbl, A_log, S, P);
  scan_phaseB<<<(B_*DIN_*NS_)/256, 256, 0, stream>>>(S, P, Ti);
  scan_phaseC<<<B_*CHUNKS*2, 256, 0, stream>>>(dlt, xi, xdbl, A_log, D_par, Ti, xr, yv);

  /* h2 = yv @ W_out + b_out + x */
  gemm16_wmma<2,4, 256,512, 512,256,256>
      <<<(NROWS/128)*(256/64), 256, 0, stream>>>(yv, W_out, h2, b_out, x);

  /* out = gelu(h2 @ W_dense + b_dense) */
  gemm16_wmma<3,2, 32,256, 256,32,32>
      <<<(NROWS/128)*(32/32), 256, 0, stream>>>(h2, W_dns, out, b_dns, nullptr);
}
